// BondPredictor_12017318494535
// MI455X (gfx1250) — matrix operand order repaired and checked by reading, not compiled, not run on hardware
//
#include <hip/hip_runtime.h>
#include <hip/hip_bf16.h>
#include <stdint.h>

typedef __attribute__((ext_vector_type(16))) _Float16 v16h;
typedef __attribute__((ext_vector_type(8)))  _Float16 v8h;
typedef __attribute__((ext_vector_type(8)))  float    v8f;

#define NT     16
#define ND     256
#define ED     128
#define TIMED  16
#define DISTD  16
#define NEDGE  5
#define FNUMT  1000.0f

#define K_EIN  672   /* 657 padded, 21 chunks of 32 */
#define K_MSG  384   /* 12 chunks */
#define K_NIN  544   /* 513 padded, 17 chunks */
#define K_EXT  384   /* 12 chunks */

// ---------------------------------------------------------------- WMMA helpers

// A fragment (16x32 f16, M on lanes): lane m holds row m; K pieces
// {8h..8h+7} and {16+8h..16+8h+7}  (ISA 7.12.2 table) -> two 16B loads.
__device__ __forceinline__ v16h load_a_frag(const _Float16* row, int half) {
  v8h lo = *(const v8h*)(row + 8 * half);
  v8h hi = *(const v8h*)(row + 16 + 8 * half);
  v16h a;
#pragma unroll
  for (int i = 0; i < 8; ++i) { a[i] = lo[i]; a[8 + i] = hi[i]; }
  return a;
}

// B fragment (32x16 f16): lane n holds column n, K pieces {8h..8h+7} and
// {16+8h..16+8h+7} (same as A). Weights pre-transposed to [N][Kpad] -> two 16B loads.
__device__ __forceinline__ v16h load_b_frag(const _Float16* wt_row_k, int half) {
  v8h lo = *(const v8h*)(wt_row_k + 8 * half);
  v8h hi = *(const v8h*)(wt_row_k + 16 + 8 * half);
  v16h b;
#pragma unroll
  for (int i = 0; i < 8; ++i) { b[i] = lo[i]; b[8 + i] = hi[i]; }
  return b;
}

__device__ __forceinline__ v8f wmma16(v16h a, v16h b, v8f c) {
  return __builtin_amdgcn_wmma_f32_16x16x32_f16(false, a, false, b,
                                                (short)0, c, false, false);
}

__device__ __forceinline__ float gsmear1(float x, float stop, int i, int n) {
  float delta = stop / (float)(n - 1);
  float off = delta * (float)i;
  float df = x - off;
  return __expf((-0.5f / (delta * delta)) * df * df);
}

// One K-chunk for a 32(M)x128(N) tile: 2 A frags share 8 B frags.
// B loads batched in groups of 4 so one wait covers 8 WMMAs.
#define CHUNK32(p0, p1, wb, kstr, acc0, acc1)                                  \
  {                                                                            \
    v16h a0 = load_a_frag((p0), half);                                         \
    v16h a1 = load_a_frag((p1), half);                                         \
    _Pragma("unroll")                                                          \
    for (int g = 0; g < 2; ++g) {                                              \
      v16h bf[4];                                                              \
      _Pragma("unroll")                                                        \
      for (int q = 0; q < 4; ++q)                                              \
        bf[q] = load_b_frag((wb) + (size_t)((g * 4 + q) * 16 + ncol) * (kstr), \
                            half);                                             \
      _Pragma("unroll")                                                        \
      for (int q = 0; q < 4; ++q) {                                            \
        acc0[g * 4 + q] = wmma16(a0, bf[q], acc0[g * 4 + q]);                  \
        acc1[g * 4 + q] = wmma16(a1, bf[q], acc1[g * 4 + q]);                  \
      }                                                                        \
    }                                                                          \
  }

// ---------------------------------------------------------------- prep kernels

__global__ void transpose_w_kernel(const float* __restrict__ src,
                                   _Float16* __restrict__ dst,
                                   int K, int Nout, int Kpad) {
  int idx = blockIdx.x * blockDim.x + threadIdx.x;
  if (idx >= Nout * Kpad) return;
  int n = idx / Kpad, k = idx - n * Kpad;
  float v = (k < K) ? src[(size_t)k * Nout + n] : 0.0f;
  dst[idx] = (_Float16)v;
}

__global__ void zero_f32_kernel(float* p, size_t n) {
  size_t i = blockIdx.x * (size_t)blockDim.x + threadIdx.x;
  if (i < n) p[i] = 0.0f;
}

__global__ void f32_to_f16_kernel(const float* __restrict__ s,
                                  _Float16* __restrict__ d, size_t n) {
  size_t i = blockIdx.x * (size_t)blockDim.x + threadIdx.x;
  if (i < n) d[i] = (_Float16)s[i];
}

__global__ __launch_bounds__(256) void embed_nodes_kernel(
    const float* __restrict__ h_node, const int* __restrict__ batch_node,
    const int* __restrict__ t, const float* __restrict__ Wne,
    float* __restrict__ hn32, _Float16* __restrict__ hn16,
    _Float16* __restrict__ ntail, int N) {
  int n = blockIdx.x;
  int j = threadIdx.x;
  __shared__ float sh[NT];
  __shared__ float stn;
  if (j < NT) sh[j] = h_node[(size_t)n * NT + j];
  if (j == 0) stn = (float)t[batch_node[n]];
  __syncthreads();
  float tn = stn;
  float v;
  if (j < ND - TIMED) {
    float s = 0.0f;
#pragma unroll
    for (int k = 0; k < NT; ++k) s += sh[k] * Wne[k * (ND - TIMED) + j];
    v = s;
  } else {
    v = gsmear1(tn, FNUMT, j - (ND - TIMED), TIMED);
  }
  hn32[(size_t)n * ND + j] = v;
  hn16[(size_t)n * ND + j] = (_Float16)v;
  if (j < 32)
    ntail[(size_t)n * 32 + j] = (j == 0) ? (_Float16)(tn / FNUMT) : (_Float16)0.0f;
}

__global__ __launch_bounds__(128) void embed_edges_kernel(
    const float* __restrict__ h_node, const float* __restrict__ pos,
    const int* __restrict__ srcv, const int* __restrict__ dstv,
    const int* __restrict__ batch_edge, const int* __restrict__ t,
    const float* __restrict__ Wee,
    float* __restrict__ he32, _Float16* __restrict__ he16,
    _Float16* __restrict__ etail, int E) {
  int e = blockIdx.x;
  int j = threadIdx.x;
  int s = srcv[e], d = dstv[e];
  float te = (float)t[batch_edge[e]];
  float v;
  if (j < ED - TIMED) {
    float acc = 0.0f;
#pragma unroll
    for (int k = 0; k < NT; ++k)
      acc += h_node[(size_t)s * NT + k] * Wee[k * (ED - TIMED) + j];
#pragma unroll
    for (int k = 0; k < NT; ++k)
      acc += h_node[(size_t)d * NT + k] * Wee[(NT + k) * (ED - TIMED) + j];
    v = acc;
  } else {
    v = gsmear1(te, FNUMT, j - (ED - TIMED), TIMED);
  }
  he32[(size_t)e * ED + j] = v;
  he16[(size_t)e * ED + j] = (_Float16)v;
  if (j < 32) {
    _Float16 tv = (_Float16)0.0f;
    if (j < DISTD) {
      float dx = pos[(size_t)d * 3 + 0] - pos[(size_t)s * 3 + 0];
      float dy = pos[(size_t)d * 3 + 1] - pos[(size_t)s * 3 + 1];
      float dz = pos[(size_t)d * 3 + 2] - pos[(size_t)s * 3 + 2];
      float dist = sqrtf(dx * dx + dy * dy + dz * dz);
      tv = (_Float16)gsmear1(dist, 10.0f, j, DISTD);
    } else if (j == DISTD) {
      tv = (_Float16)(te / FNUMT);
    }
    etail[(size_t)e * 32 + j] = tv;
  }
}

// ---------------------------------------------------------------- GEMM kernels

// h_e += relu(e_in @ We1 + be1) @ We2 + be2 ; e_in = [h_e|hn[src]|hn[dst]|dist|time]
__global__ __launch_bounds__(64) void edge_mlp_kernel(
    const int* __restrict__ srcv, const int* __restrict__ dstv,
    const _Float16* __restrict__ hn16, _Float16* __restrict__ he16,
    float* __restrict__ he32, const _Float16* __restrict__ etail,
    const _Float16* __restrict__ W1t, const float* __restrict__ b1,
    const _Float16* __restrict__ W2t, const float* __restrict__ b2, int E) {
  __shared__ alignas(32) _Float16 lds_hid[2][32][ED];
  int wave = threadIdx.x >> 5, lane = threadIdx.x & 31;
  int half = lane >> 4, ncol = lane & 15;
  int strips = (E + 31) >> 5;
  int strip = blockIdx.x * 2 + wave;
  bool active = strip < strips;
  if (!active) strip = strips - 1;
  int m0 = strip << 5;
  int mr0 = m0 + (lane & 15);       if (mr0 >= E) mr0 = E - 1;
  int mr1 = m0 + 16 + (lane & 15);  if (mr1 >= E) mr1 = E - 1;

  int s0 = srcv[mr0], d0 = dstv[mr0];
  int s1 = srcv[mr1], d1 = dstv[mr1];
  const _Float16* rHE0 = he16 + (size_t)mr0 * ED;
  const _Float16* rHE1 = he16 + (size_t)mr1 * ED;
  const _Float16* rS0  = hn16 + (size_t)s0 * ND;
  const _Float16* rS1  = hn16 + (size_t)s1 * ND;
  const _Float16* rD0  = hn16 + (size_t)d0 * ND;
  const _Float16* rD1  = hn16 + (size_t)d1 * ND;
  const _Float16* rT0  = etail + (size_t)mr0 * 32;
  const _Float16* rT1  = etail + (size_t)mr1 * 32;

  v8f acc0[8] = {}, acc1[8] = {};
#pragma unroll 1
  for (int c = 0; c < 4; ++c)
    CHUNK32(rHE0 + c * 32, rHE1 + c * 32, W1t + c * 32, K_EIN, acc0, acc1);
#pragma unroll 1
  for (int c = 0; c < 8; ++c)
    CHUNK32(rS0 + c * 32, rS1 + c * 32, W1t + (4 + c) * 32, K_EIN, acc0, acc1);
#pragma unroll 1
  for (int c = 0; c < 8; ++c)
    CHUNK32(rD0 + c * 32, rD1 + c * 32, W1t + (12 + c) * 32, K_EIN, acc0, acc1);
  CHUNK32(rT0, rT1, W1t + 20 * 32, K_EIN, acc0, acc1);

#pragma unroll
  for (int nt = 0; nt < 8; ++nt) {
    int n = nt * 16 + ncol;
    float bb = b1[n];
#pragma unroll
    for (int j = 0; j < 8; ++j) {
      float v0 = acc0[nt][j] + bb;
      float v1 = acc1[nt][j] + bb;
      lds_hid[wave][j + 8 * half][n]      = (_Float16)(v0 > 0.0f ? v0 : 0.0f);
      lds_hid[wave][16 + j + 8 * half][n] = (_Float16)(v1 > 0.0f ? v1 : 0.0f);
    }
  }
  __syncthreads();
  const _Float16* rowH0 = &lds_hid[wave][lane & 15][0];
  const _Float16* rowH1 = &lds_hid[wave][16 + (lane & 15)][0];
  v8f acd0[8] = {}, acd1[8] = {};
#pragma unroll 1
  for (int c = 0; c < 4; ++c)
    CHUNK32(rowH0 + c * 32, rowH1 + c * 32, W2t + c * 32, ED, acd0, acd1);

#pragma unroll
  for (int nt = 0; nt < 8; ++nt) {
    int n = nt * 16 + ncol;
    float bb = b2[n];
#pragma unroll
    for (int j = 0; j < 8; ++j) {
      int ma = m0 + j + 8 * half;
      int mb = ma + 16;
      if (active && ma < E) {
        size_t off = (size_t)ma * ED + n;
        float v = he32[off] + acd0[nt][j] + bb;
        he32[off] = v;
        he16[off] = (_Float16)v;
      }
      if (active && mb < E) {
        size_t off = (size_t)mb * ED + n;
        float v = he32[off] + acd1[nt][j] + bb;
        he32[off] = v;
        he16[off] = (_Float16)v;
      }
    }
  }
}

// msg = relu([hn[src]|h_e] @ Wm + bm) ; agg[dst] += msg
__global__ __launch_bounds__(64) void msg_agg_kernel(
    const int* __restrict__ srcv, const int* __restrict__ dstv,
    const _Float16* __restrict__ hn16, const _Float16* __restrict__ he16,
    const _Float16* __restrict__ Wmt, const float* __restrict__ bm,
    float* __restrict__ agg, int E) {
  int wave = threadIdx.x >> 5, lane = threadIdx.x & 31;
  int half = lane >> 4, ncol = lane & 15;
  int strips = (E + 31) >> 5;
  int strip = blockIdx.x * 2 + wave;
  bool active = strip < strips;
  if (!active) strip = strips - 1;
  int m0 = strip << 5;
  int mr0 = m0 + (lane & 15);       if (mr0 >= E) mr0 = E - 1;
  int mr1 = m0 + 16 + (lane & 15);  if (mr1 >= E) mr1 = E - 1;
  int s0 = srcv[mr0], s1 = srcv[mr1];
  const _Float16* rS0  = hn16 + (size_t)s0 * ND;
  const _Float16* rS1  = hn16 + (size_t)s1 * ND;
  const _Float16* rHE0 = he16 + (size_t)mr0 * ED;
  const _Float16* rHE1 = he16 + (size_t)mr1 * ED;

#pragma unroll 1
  for (int nh = 0; nh < 2; ++nh) {
    const _Float16* wbase = Wmt + (size_t)(nh * 128) * K_MSG;
    v8f acc0[8] = {}, acc1[8] = {};
#pragma unroll 1
    for (int c = 0; c < 8; ++c)
      CHUNK32(rS0 + c * 32, rS1 + c * 32, wbase + c * 32, K_MSG, acc0, acc1);
#pragma unroll 1
    for (int c = 0; c < 4; ++c)
      CHUNK32(rHE0 + c * 32, rHE1 + c * 32, wbase + (8 + c) * 32, K_MSG, acc0, acc1);

#pragma unroll
    for (int nt = 0; nt < 8; ++nt) {
      int n = nh * 128 + nt * 16 + ncol;
      float bb = bm[n];
#pragma unroll
      for (int j = 0; j < 8; ++j) {
        int ma = m0 + j + 8 * half;
        int mb = ma + 16;
        if (active && ma < E) {
          float v = acc0[nt][j] + bb;
          v = v > 0.0f ? v : 0.0f;
          atomicAdd(&agg[(size_t)dstv[ma] * ND + n], v);
        }
        if (active && mb < E) {
          float v = acc1[nt][j] + bb;
          v = v > 0.0f ? v : 0.0f;
          atomicAdd(&agg[(size_t)dstv[mb] * ND + n], v);
        }
      }
    }
  }
}

// h_n += relu([h_n|agg|time] @ Wn1 + bn1) @ Wn2 + bn2
__global__ __launch_bounds__(64) void node_mlp_kernel(
    const _Float16* __restrict__ agg16, const _Float16* __restrict__ ntail,
    _Float16* __restrict__ hn16, float* __restrict__ hn32,
    const _Float16* __restrict__ W1t, const float* __restrict__ b1,
    const _Float16* __restrict__ W2t, const float* __restrict__ b2, int N) {
  __shared__ alignas(32) _Float16 lds_hid[2][32][ND];
  int wave = threadIdx.x >> 5, lane = threadIdx.x & 31;
  int half = lane >> 4, ncol = lane & 15;
  int strips = (N + 31) >> 5;
  int strip = blockIdx.x * 2 + wave;
  bool active = strip < strips;
  if (!active) strip = strips - 1;
  int m0 = strip << 5;
  int mr0 = m0 + (lane & 15);       if (mr0 >= N) mr0 = N - 1;
  int mr1 = m0 + 16 + (lane & 15);  if (mr1 >= N) mr1 = N - 1;
  const _Float16* rN0 = hn16 + (size_t)mr0 * ND;
  const _Float16* rN1 = hn16 + (size_t)mr1 * ND;
  const _Float16* rA0 = agg16 + (size_t)mr0 * ND;
  const _Float16* rA1 = agg16 + (size_t)mr1 * ND;
  const _Float16* rT0 = ntail + (size_t)mr0 * 32;
  const _Float16* rT1 = ntail + (size_t)mr1 * 32;

#pragma unroll 1
  for (int nh = 0; nh < 2; ++nh) {
    const _Float16* wbase = W1t + (size_t)(nh * 128) * K_NIN;
    v8f acc0[8] = {}, acc1[8] = {};
#pragma unroll 1
    for (int c = 0; c < 8; ++c)
      CHUNK32(rN0 + c * 32, rN1 + c * 32, wbase + c * 32, K_NIN, acc0, acc1);
#pragma unroll 1
    for (int c = 0; c < 8; ++c)
      CHUNK32(rA0 + c * 32, rA1 + c * 32, wbase + (8 + c) * 32, K_NIN, acc0, acc1);
    CHUNK32(rT0, rT1, wbase + 16 * 32, K_NIN, acc0, acc1);

#pragma unroll
    for (int nt = 0; nt < 8; ++nt) {
      int n = nh * 128 + nt * 16 + ncol;
      float bb = b1[n];
#pragma unroll
      for (int j = 0; j < 8; ++j) {
        float v0 = acc0[nt][j] + bb;
        float v1 = acc1[nt][j] + bb;
        lds_hid[wave][j + 8 * half][n]      = (_Float16)(v0 > 0.0f ? v0 : 0.0f);
        lds_hid[wave][16 + j + 8 * half][n] = (_Float16)(v1 > 0.0f ? v1 : 0.0f);
      }
    }
  }
  __syncthreads();
  const _Float16* rowH0 = &lds_hid[wave][lane & 15][0];
  const _Float16* rowH1 = &lds_hid[wave][16 + (lane & 15)][0];
#pragma unroll 1
  for (int nh = 0; nh < 2; ++nh) {
    const _Float16* wbase = W2t + (size_t)(nh * 128) * ND;
    v8f acc0[8] = {}, acc1[8] = {};
#pragma unroll 1
    for (int c = 0; c < 8; ++c)
      CHUNK32(rowH0 + c * 32, rowH1 + c * 32, wbase + c * 32, ND, acc0, acc1);

#pragma unroll
    for (int nt = 0; nt < 8; ++nt) {
      int n = nh * 128 + nt * 16 + ncol;
      float bb = b2[n];
#pragma unroll
      for (int j = 0; j < 8; ++j) {
        int ma = m0 + j + 8 * half;
        int mb = ma + 16;
        if (active && ma < N) {
          size_t off = (size_t)ma * ND + n;
          float v = hn32[off] + acc0[nt][j] + bb;
          hn32[off] = v;
          hn16[off] = (_Float16)v;
        }
        if (active && mb < N) {
          size_t off = (size_t)mb * ND + n;
          float v = hn32[off] + acc1[nt][j] + bb;
          hn32[off] = v;
          hn16[off] = (_Float16)v;
        }
      }
    }
  }
}

__global__ void build_hext_kernel(const float* __restrict__ he32,
                                  const float* __restrict__ hn32,
                                  const int* __restrict__ srcv,
                                  const int* __restrict__ dstv,
                                  _Float16* __restrict__ hext, int EH) {
  size_t i = blockIdx.x * (size_t)blockDim.x + threadIdx.x;
  size_t total = (size_t)EH * K_EXT;
  if (i >= total) return;
  int e = (int)(i / K_EXT), k = (int)(i - (size_t)e * K_EXT);
  float v;
  if (k < ED) {
    v = he32[(size_t)e * ED + k] + he32[(size_t)(e + EH) * ED + k];
  } else {
    int s = srcv[e], d = dstv[e];
    v = hn32[(size_t)s * ND + (k - ED)] + hn32[(size_t)d * ND + (k - ED)];
  }
  hext[i] = (_Float16)v;
}

// out = relu(relu(hext@Wd1+bd1)@Wd2+bd2)@Wd3+bd3
__global__ __launch_bounds__(64) void decoder_kernel(
    const _Float16* __restrict__ hext,
    const _Float16* __restrict__ W1t, const float* __restrict__ b1,
    const _Float16* __restrict__ W2t, const float* __restrict__ b2,
    const float* __restrict__ W3, const float* __restrict__ b3,
    float* __restrict__ out, int EH) {
  __shared__ alignas(32) _Float16 lds_hid[2][32][ED];
  __shared__ float lds_h2[2][32][ED];
  int wave = threadIdx.x >> 5, lane = threadIdx.x & 31;
  int half = lane >> 4, ncol = lane & 15;
  int strips = (EH + 31) >> 5;
  int strip = blockIdx.x * 2 + wave;
  bool active = strip < strips;
  if (!active) strip = strips - 1;
  int m0 = strip << 5;
  int mr0 = m0 + (lane & 15);       if (mr0 >= EH) mr0 = EH - 1;
  int mr1 = m0 + 16 + (lane & 15);  if (mr1 >= EH) mr1 = EH - 1;
  const _Float16* rX0 = hext + (size_t)mr0 * K_EXT;
  const _Float16* rX1 = hext + (size_t)mr1 * K_EXT;

  v8f acc0[8] = {}, acc1[8] = {};
#pragma unroll 1
  for (int c = 0; c < 12; ++c)
    CHUNK32(rX0 + c * 32, rX1 + c * 32, W1t + c * 32, K_EXT, acc0, acc1);
#pragma unroll
  for (int nt = 0; nt < 8; ++nt) {
    int n = nt * 16 + ncol;
    float bb = b1[n];
#pragma unroll
    for (int j = 0; j < 8; ++j) {
      float v0 = acc0[nt][j] + bb;
      float v1 = acc1[nt][j] + bb;
      lds_hid[wave][j + 8 * half][n]      = (_Float16)(v0 > 0.0f ? v0 : 0.0f);
      lds_hid[wave][16 + j + 8 * half][n] = (_Float16)(v1 > 0.0f ? v1 : 0.0f);
    }
  }
  __syncthreads();
  const _Float16* rowH0 = &lds_hid[wave][lane & 15][0];
  const _Float16* rowH1 = &lds_hid[wave][16 + (lane & 15)][0];
  v8f acd0[8] = {}, acd1[8] = {};
#pragma unroll 1
  for (int c = 0; c < 4; ++c)
    CHUNK32(rowH0 + c * 32, rowH1 + c * 32, W2t + c * 32, ED, acd0, acd1);
#pragma unroll
  for (int nt = 0; nt < 8; ++nt) {
    int n = nt * 16 + ncol;
    float bb = b2[n];
#pragma unroll
    for (int j = 0; j < 8; ++j) {
      float v0 = acd0[nt][j] + bb;
      float v1 = acd1[nt][j] + bb;
      lds_h2[wave][j + 8 * half][n]      = v0 > 0.0f ? v0 : 0.0f;
      lds_h2[wave][16 + j + 8 * half][n] = v1 > 0.0f ? v1 : 0.0f;
    }
  }
  __syncthreads();
  for (int idx = lane; idx < 32 * NEDGE; idx += 32) {
    int m = idx / NEDGE, c = idx - m * NEDGE;
    float sum = b3[c];
    for (int k = 0; k < ED; ++k) sum += lds_h2[wave][m][k] * W3[k * NEDGE + c];
    if (active && (m0 + m) < EH) out[(size_t)(m0 + m) * NEDGE + c] = sum;
  }
}

// ---------------------------------------------------------------- host

extern "C" void kernel_launch(void* const* d_in, const int* in_sizes, int n_in,
                              void* d_out, int out_size, void* d_ws, size_t ws_size,
                              hipStream_t stream) {
  const float* h_node     = (const float*)d_in[0];
  const float* pos        = (const float*)d_in[1];
  const int*   batch_node = (const int*)d_in[2];
  const int*   edge_index = (const int*)d_in[3];
  const int*   batch_edge = (const int*)d_in[4];
  const int*   t          = (const int*)d_in[5];
  const float* W_node_emb = (const float*)d_in[6];
  const float* W_edge_emb = (const float*)d_in[7];
  const float* We1 = (const float*)d_in[8];
  const float* be1 = (const float*)d_in[9];
  const float* We2 = (const float*)d_in[10];
  const float* be2 = (const float*)d_in[11];
  const float* Wm  = (const float*)d_in[12];
  const float* bm  = (const float*)d_in[13];
  const float* Wn1 = (const float*)d_in[14];
  const float* bn1 = (const float*)d_in[15];
  const float* Wn2 = (const float*)d_in[16];
  const float* bn2 = (const float*)d_in[17];
  const float* Wd1 = (const float*)d_in[18];
  const float* bd1 = (const float*)d_in[19];
  const float* Wd2 = (const float*)d_in[20];
  const float* bd2 = (const float*)d_in[21];
  const float* Wd3 = (const float*)d_in[22];
  const float* bd3 = (const float*)d_in[23];

  const int N  = in_sizes[0] / NT;
  const int E  = in_sizes[4];
  const int EH = E / 2;
  const int* srcv = edge_index;
  const int* dstv = edge_index + E;

  char* ws = (char*)d_ws;
  size_t off = 0;
  auto alloc = [&](size_t bytes) -> void* {
    off = (off + 255) & ~(size_t)255;
    void* p = ws + off;
    off += bytes;
    return p;
  };

  float*     hn32  = (float*)    alloc((size_t)N * ND * 4);
  _Float16*  hn16  = (_Float16*) alloc((size_t)N * ND * 2);
  _Float16*  ntail = (_Float16*) alloc((size_t)N * 32 * 2);
  float*     he32  = (float*)    alloc((size_t)E * ED * 4);
  _Float16*  he16  = (_Float16*) alloc((size_t)E * ED * 2);
  _Float16*  etail = (_Float16*) alloc((size_t)E * 32 * 2);
  float*     agg32 = (float*)    alloc((size_t)N * ND * 4);
  _Float16*  agg16 = (_Float16*) alloc((size_t)N * ND * 2);
  _Float16*  hext  = (_Float16*) alloc((size_t)EH * K_EXT * 2);
  _Float16*  We1t  = (_Float16*) alloc((size_t)2 * ED * K_EIN * 2);
  _Float16*  We2t  = (_Float16*) alloc((size_t)2 * ED * ED * 2);
  _Float16*  Wmt   = (_Float16*) alloc((size_t)2 * ND * K_MSG * 2);
  _Float16*  Wn1t  = (_Float16*) alloc((size_t)2 * ND * K_NIN * 2);
  _Float16*  Wn2t  = (_Float16*) alloc((size_t)2 * ND * ND * 2);
  _Float16*  Wd1t  = (_Float16*) alloc((size_t)ED * K_EXT * 2);
  _Float16*  Wd2t  = (_Float16*) alloc((size_t)ED * ED * 2);
  (void)ws_size; (void)n_in; (void)out_size;

  auto tr = [&](const float* src, _Float16* dst, int K, int Nout, int Kpad) {
    int total = Nout * Kpad;
    transpose_w_kernel<<<(total + 255) / 256, 256, 0, stream>>>(src, dst, K, Nout, Kpad);
  };
  for (int l = 0; l < 2; ++l) {
    tr(We1 + (size_t)l * 657 * ED, We1t + (size_t)l * ED * K_EIN, 657, ED, K_EIN);
    tr(We2 + (size_t)l * ED * ED,  We2t + (size_t)l * ED * ED,   ED,  ED, ED);
    tr(Wm  + (size_t)l * 384 * ND, Wmt  + (size_t)l * ND * K_MSG, 384, ND, K_MSG);
    tr(Wn1 + (size_t)l * 513 * ND, Wn1t + (size_t)l * ND * K_NIN, 513, ND, K_NIN);
    tr(Wn2 + (size_t)l * ND * ND,  Wn2t + (size_t)l * ND * ND,   ND,  ND, ND);
  }
  tr(Wd1, Wd1t, 384, ED, K_EXT);
  tr(Wd2, Wd2t, ED, ED, ED);

  embed_nodes_kernel<<<N, 256, 0, stream>>>(h_node, batch_node, t, W_node_emb,
                                            hn32, hn16, ntail, N);
  embed_edges_kernel<<<E, 128, 0, stream>>>(h_node, pos, srcv, dstv, batch_edge,
                                            t, W_edge_emb, he32, he16, etail, E);

  const int estrips = (E + 31) / 32, eblocks = (estrips + 1) / 2;
  const int nstrips = (N + 31) / 32, nblocks = (nstrips + 1) / 2;
  const int dstrips = (EH + 31) / 32, dblocks = (dstrips + 1) / 2;
  const size_t an = (size_t)N * ND;

  for (int l = 0; l < 2; ++l) {
    edge_mlp_kernel<<<eblocks, 64, 0, stream>>>(
        srcv, dstv, hn16, he16, he32, etail,
        We1t + (size_t)l * ED * K_EIN, be1 + (size_t)l * ED,
        We2t + (size_t)l * ED * ED,   be2 + (size_t)l * ED, E);
    zero_f32_kernel<<<(int)((an + 255) / 256), 256, 0, stream>>>(agg32, an);
    msg_agg_kernel<<<eblocks, 64, 0, stream>>>(
        srcv, dstv, hn16, he16,
        Wmt + (size_t)l * ND * K_MSG, bm + (size_t)l * ND, agg32, E);
    f32_to_f16_kernel<<<(int)((an + 255) / 256), 256, 0, stream>>>(agg32, agg16, an);
    node_mlp_kernel<<<nblocks, 64, 0, stream>>>(
        agg16, ntail, hn16, hn32,
        Wn1t + (size_t)l * ND * K_NIN, bn1 + (size_t)l * ND,
        Wn2t + (size_t)l * ND * ND,   bn2 + (size_t)l * ND, N);
  }

  size_t ht = (size_t)EH * K_EXT;
  build_hext_kernel<<<(int)((ht + 255) / 256), 256, 0, stream>>>(he32, hn32, srcv,
                                                                 dstv, hext, EH);
  decoder_kernel<<<dblocks, 64, 0, stream>>>(hext, Wd1t, bd1, Wd2t, bd2, Wd3,
                                             bd3, (float*)d_out, EH);
}
